// SupConLossDis_54150947668039
// MI455X (gfx1250) — matrix operand order repaired and checked by reading, not compiled, not run on hardware
//
#include <hip/hip_runtime.h>

// ---------------------------------------------------------------------------
// SupConLoss (distance form) for features (4096, 2, 128) f32 -> scalar f32.
// Gram GEMM with v_wmma_f32_16x16x32_f16 (f16 in, f32 accumulate); softmax
// fused into the GEMM epilogue (the logits_max shift cancels analytically).
// Compute-bound: 17.2 GFLOP Gram vs 4 MB input -> WMMA throughput dominates.
// ---------------------------------------------------------------------------

typedef _Float16 half4v __attribute__((ext_vector_type(4)));
typedef _Float16 half8v __attribute__((ext_vector_type(8)));
typedef _Float16 v16h   __attribute__((ext_vector_type(16)));
typedef float    v8f    __attribute__((ext_vector_type(8)));

#define NROWS            8192
#define DIM              128
#define BSZ              4096
#define TILE             16
#define NTILES           (NROWS / TILE)            // 512 column tiles
#define WAVES_PER_WG     8
#define ROWS_PER_WG      (WAVES_PER_WG * TILE)     // 128
#define ROW_WGS          (NROWS / ROWS_PER_WG)     // 64
#define COL_CHUNKS       4
#define TILES_PER_CHUNK  (NTILES / COL_CHUNKS)     // 128
#define CTILE            32                        // columns staged per iteration
#define ITERS            (TILES_PER_CHUNK / 2)     // 64 iterations per chunk
#define LDS_ROW          136                       // 128 halves + 8 pad (272 B)

static __device__ __forceinline__ v16h cat8(half8v lo, half8v hi) {
  return __builtin_shufflevector(lo, hi, 0,1,2,3,4,5,6,7,8,9,10,11,12,13,14,15);
}

// ---------------------------------------------------------------------------
// Kernel 1: view-major restack + L2 row normalize, emit f16 table Xh[N][D].
// ---------------------------------------------------------------------------
__global__ void __launch_bounds__(256) normalize_rows(const float* __restrict__ feat,
                                                      _Float16* __restrict__ Xh) {
  const int row  = (blockIdx.x * blockDim.x + threadIdx.x) >> 5;
  const int lane = threadIdx.x & 31;
  if (row >= NROWS) return;
  const int b = row & (BSZ - 1);      // original sample
  const int v = row >> 12;            // view index (after swapaxes stacking)
  const float4 x = ((const float4*)(feat + (size_t)(b * 2 + v) * DIM))[lane];
  float ss = x.x * x.x + x.y * x.y + x.z * x.z + x.w * x.w;
#pragma unroll
  for (int m = 16; m >= 1; m >>= 1) ss += __shfl_xor(ss, m, 32);
  const float inv = 1.0f / sqrtf(ss);
  half4v h;
  h[0] = (_Float16)(x.x * inv);
  h[1] = (_Float16)(x.y * inv);
  h[2] = (_Float16)(x.z * inv);
  h[3] = (_Float16)(x.w * inv);
  ((half4v*)(Xh + (size_t)row * DIM))[lane] = h;
}

// ---------------------------------------------------------------------------
// Kernel 2: Gram tile GEMM + fused (unshifted) softmax statistics.
// Wave owns a 16-row strip; A-fragments live in VGPRs for the whole kernel.
// Per iteration: 32 columns double-buffered through LDS, all B fragments
// loaded up front, 8 back-to-back WMMAs on two accumulators, then a
// wave-uniform fast/slow epilogue with no long-lived temporaries.
// ---------------------------------------------------------------------------
__global__ void __launch_bounds__(256, 1) gram_softmax(const _Float16* __restrict__ Xh,
                                                       float* __restrict__ rowsum,
                                                       float* __restrict__ prow) {
  __shared__ _Float16 lds[2][CTILE * LDS_ROW];    // 2 x 8.5 KB

  const int tid  = threadIdx.x;
  const int lane = tid & 31;
  const int wv   = tid >> 5;          // wave 0..7
  const int hi   = lane >> 4;         // half-wave
  const int l16  = lane & 15;
  const int row_base  = blockIdx.x * ROWS_PER_WG + wv * TILE;
  const int part_base = row_base ^ BSZ;                       // partner-row strip
  const int col0      = blockIdx.y * TILES_PER_CHUNK * TILE;  // chunk column base

  // A fragments (16-bit A 16x32): lane<16 -> K {0..7,16..23}, lane>=16 -> K {8..15,24..31}
  v16h afrag[4];
  {
    const _Float16* aptr = Xh + (size_t)(row_base + l16) * DIM;
#pragma unroll
    for (int s = 0; s < 4; ++s) {
      half8v lo = *(const half8v*)(aptr + 32 * s + hi * 8);
      half8v hh = *(const half8v*)(aptr + 32 * s + 16 + hi * 8);
      afrag[s] = cat8(lo, hh);
    }
  }

  float sumexp[8], psum[8];
#pragma unroll
  for (int r = 0; r < 8; ++r) { sumexp[r] = 0.0f; psum[r] = 0.0f; }

  // cooperative staging: 512 x 16B segments per iteration, 2 per thread
  const int r0 = tid >> 4,          g0 = tid & 15;          // segment 0
  const int r1 = (tid + 256) >> 4,  g1 = tid & 15;          // segment 1 (rows 16..31)

  { // preload iteration 0
    half8v a0 = *(const half8v*)(Xh + (size_t)(col0 + r0) * DIM + g0 * 8);
    half8v a1 = *(const half8v*)(Xh + (size_t)(col0 + r1) * DIM + g1 * 8);
    *(half8v*)&lds[0][r0 * LDS_ROW + g0 * 8] = a0;
    *(half8v*)&lds[0][r1 * LDS_ROW + g1 * 8] = a1;
  }
  __syncthreads();

  for (int it = 0; it < ITERS; ++it) {
    const int p = it & 1;
    if (it + 1 < ITERS) {
      const int nb = col0 + (it + 1) * CTILE;
      half8v a0 = *(const half8v*)(Xh + (size_t)(nb + r0) * DIM + g0 * 8);
      half8v a1 = *(const half8v*)(Xh + (size_t)(nb + r1) * DIM + g1 * 8);
      *(half8v*)&lds[p ^ 1][r0 * LDS_ROW + g0 * 8] = a0;
      *(half8v*)&lds[p ^ 1][r1 * LDS_ROW + g1 * 8] = a1;
      if (it + 2 < ITERS)   // WGP-scope prefetch of the upcoming column stream
        __builtin_prefetch((const void*)(Xh + (size_t)(col0 + (it + 2) * CTILE) * DIM), 0, 3);
    }

    // B fragments (32x16): lane holds column l16, local K {hi*8..hi*8+7, 16+hi*8..16+hi*8+7}.
    // Load ALL fragments first so the dscnt waits cover batches and the
    // 8 WMMAs issue back-to-back.
    const _Float16* b0 = &lds[p][l16 * LDS_ROW + hi * 8];
    const _Float16* b1 = b0 + 16 * LDS_ROW;
    v16h bf0[4], bf1[4];
#pragma unroll
    for (int s = 0; s < 4; ++s) {
      bf0[s] = cat8(*(const half8v*)(b0 + 32 * s), *(const half8v*)(b0 + 32 * s + 16));
      bf1[s] = cat8(*(const half8v*)(b1 + 32 * s), *(const half8v*)(b1 + 32 * s + 16));
    }

    v8f c0 = {}, c1 = {};
#pragma unroll
    for (int s = 0; s < 4; ++s)
      c0 = __builtin_amdgcn_wmma_f32_16x16x32_f16(false, afrag[s], false, bf0[s],
                                                  (short)0, c0, false, false);
#pragma unroll
    for (int s = 0; s < 4; ++s)
      c1 = __builtin_amdgcn_wmma_f32_16x16x32_f16(false, afrag[s], false, bf1[s],
                                                  (short)0, c1, false, false);

    // fused epilogue: adc = 1 - sqrt(max(2-2g, eps)); raw v_sqrt is exact
    // enough on the clamped range [1e-12, 4]. The diag/partner test is
    // wave-uniform -> scalar branch; EXEC stays all-ones around the WMMAs.
    const int cwin = col0 + it * CTILE;          // column window [cwin, cwin+32)
    const bool special = ((row_base  < cwin + CTILE) && (cwin < row_base  + TILE)) ||
                         ((part_base < cwin + CTILE) && (cwin < part_base + TILE));
    if (!special) {
      // fast path (63 or 64 of 64 iterations): no masking, no long-lived temporaries
#pragma unroll
      for (int r = 0; r < 8; ++r) {
        const float tA = fmaxf(2.0f - 2.0f * c0[r], 1e-12f);
        const float tB = fmaxf(2.0f - 2.0f * c1[r], 1e-12f);
        sumexp[r] += __expf(1.0f - __builtin_amdgcn_sqrtf(tA))
                   + __expf(1.0f - __builtin_amdgcn_sqrtf(tB));
      }
    } else {
      const int colA = cwin + l16;
      const int colB = colA + 16;
#pragma unroll
      for (int r = 0; r < 8; ++r) {
        const int gi = row_base + hi * 8 + r;    // C layout: M = r + 8*hi, N = l16
        const float tA   = fmaxf(2.0f - 2.0f * c0[r], 1e-12f);
        const float tB   = fmaxf(2.0f - 2.0f * c1[r], 1e-12f);
        const float adcA = 1.0f - __builtin_amdgcn_sqrtf(tA);
        const float adcB = 1.0f - __builtin_amdgcn_sqrtf(tB);
        const float eA   = __expf(adcA);
        const float eB   = __expf(adcB);
        sumexp[r] += (colA != gi) ? eA : 0.0f;              // exclude self
        sumexp[r] += (colB != gi) ? eB : 0.0f;
        psum[r]   += (colA == (gi ^ BSZ)) ? adcA : 0.0f;    // partner positive
        psum[r]   += (colB == (gi ^ BSZ)) ? adcB : 0.0f;
      }
    }
    __syncthreads();
  }

  // butterfly reduce across the 16 lanes of each half-wave (row-sum over N)
#pragma unroll
  for (int m = 1; m <= 8; m <<= 1) {
#pragma unroll
    for (int r = 0; r < 8; ++r) {
      sumexp[r] += __shfl_xor(sumexp[r], m, 32);
      psum[r]   += __shfl_xor(psum[r],   m, 32);
    }
  }

  // lanes l16==r publish row (row_base + hi*8 + r) for this column chunk
#pragma unroll
  for (int r = 0; r < 8; ++r) {
    if (l16 == r) {
      const int grow = row_base + hi * 8 + r;
      rowsum[blockIdx.y * NROWS + grow] = sumexp[r];
      prow  [blockIdx.y * NROWS + grow] = psum[r];
    }
  }
}

// ---------------------------------------------------------------------------
// Kernel 3: combine column chunks, loss_i = log(S_i) - adc_pos_i, mean over N.
// Single block -> deterministic (no atomics).
// ---------------------------------------------------------------------------
__global__ void __launch_bounds__(256) finalize(const float* __restrict__ rowsum,
                                                const float* __restrict__ prow,
                                                float* __restrict__ out) {
  __shared__ float red[256];
  const int t = threadIdx.x;
  float acc = 0.0f;
  for (int row = t; row < NROWS; row += 256) {
    float s = 0.0f, pp = 0.0f;
#pragma unroll
    for (int c = 0; c < COL_CHUNKS; ++c) {
      s  += rowsum[c * NROWS + row];
      pp += prow  [c * NROWS + row];
    }
    acc += __logf(s) - pp;   // -(log_prob of the positive); T/BT == 1
  }
  red[t] = acc;
  __syncthreads();
  for (int off = 128; off > 0; off >>= 1) {
    if (t < off) red[t] += red[t + off];
    __syncthreads();
  }
  if (t == 0) out[0] = red[0] / (float)NROWS;
}

// ---------------------------------------------------------------------------
extern "C" void kernel_launch(void* const* d_in, const int* in_sizes, int n_in,
                              void* d_out, int out_size, void* d_ws, size_t ws_size,
                              hipStream_t stream) {
  const float* feat = (const float*)d_in[0];

  // workspace layout: Xh (2 MB f16) | rowsum (4*8192 f32) | prow (4*8192 f32)
  _Float16* Xh     = (_Float16*)d_ws;
  float*    rowsum = (float*)((char*)d_ws + (size_t)NROWS * DIM * sizeof(_Float16));
  float*    prow   = rowsum + COL_CHUNKS * NROWS;
  float*    out    = (float*)d_out;

  normalize_rows<<<NROWS / 8, 256, 0, stream>>>(feat, Xh);
  dim3 grid(ROW_WGS, COL_CHUNKS);
  gram_softmax<<<grid, 256, 0, stream>>>(Xh, rowsum, prow);
  finalize<<<1, 256, 0, stream>>>(rowsum, prow, out);
}
